// MoERef_11716670783494
// MI455X (gfx1250) — matrix operand order repaired and checked by reading, not compiled, not run on hardware
//
#include <hip/hip_runtime.h>
#include <hip/hip_bf16.h>

// ---------------- problem constants ----------------
#define NTOK 4096
#define TOPK 2
#define NEXP 8
#define HD   2048
#define ID   1408

// ---------------- tiling ----------------
#define BM   64                      // rows per block tile (padded per-expert segments)
#define BN   128                     // cols per block tile (I for gemm1, H for gemm2)
#define KC   32                      // k-slab per stage
#define CAP  (NTOK*TOPK + NEXP*BM)   // 8704 padded pair-slots
#define MAXT (NTOK*TOPK/BM + NEXP)   // 136 max row tiles

// LDS strides/sizes (elements); stride 40 bf16 = 80B keeps 16B alignment, avoids bank conflicts
#define SAS  40
#define LDSA (BM*SAS)                // 2560 elems = 5120 B per buffer
#define LDSB (BN*SAS)                // 5120 elems = 10240 B per buffer

// ---------------- workspace layout (bytes) ----------------
#define OFF_COUNTS 0
#define OFF_CURSOR (NEXP*4)
#define OFF_OFFS   (2*NEXP*4)
#define OFF_NT     (OFF_OFFS + (NEXP+1)*4)
#define OFF_TILES  256
#define OFF_PERM   2048
#define OFF_PW     (OFF_PERM + CAP*4)            // 36864
#define OFF_HID    (OFF_PW + CAP*4)              // 71680 (256-aligned)
// hidden: CAP * ID * 2 bytes ~= 24.5 MB

typedef __bf16 bf16_t;
typedef __attribute__((ext_vector_type(16))) __bf16 v16bf;
typedef __attribute__((ext_vector_type(8)))  __bf16 v8bf;
typedef __attribute__((ext_vector_type(8)))  float  v8f;

// ---------------- converters / fragment loaders ----------------
static __device__ __forceinline__ v8bf cvt8(float4 f0, float4 f1) {
  v8bf r;
  r[0]=(bf16_t)f0.x; r[1]=(bf16_t)f0.y; r[2]=(bf16_t)f0.z; r[3]=(bf16_t)f0.w;
  r[4]=(bf16_t)f1.x; r[5]=(bf16_t)f1.y; r[6]=(bf16_t)f1.z; r[7]=(bf16_t)f1.w;
  return r;
}
static __device__ __forceinline__ v16bf cvt16(float4 f0, float4 f1, float4 f2, float4 f3) {
  v16bf r;
  r[0]=(bf16_t)f0.x;  r[1]=(bf16_t)f0.y;  r[2]=(bf16_t)f0.z;  r[3]=(bf16_t)f0.w;
  r[4]=(bf16_t)f1.x;  r[5]=(bf16_t)f1.y;  r[6]=(bf16_t)f1.z;  r[7]=(bf16_t)f1.w;
  r[8]=(bf16_t)f2.x;  r[9]=(bf16_t)f2.y;  r[10]=(bf16_t)f2.z; r[11]=(bf16_t)f2.w;
  r[12]=(bf16_t)f3.x; r[13]=(bf16_t)f3.y; r[14]=(bf16_t)f3.z; r[15]=(bf16_t)f3.w;
  return r;
}
// combine two v8bf (A and B frags: halves 8 elems apart)
static __device__ __forceinline__ v16bf join16(v8bf lo, v8bf hi) {
  v16bf r;
#pragma unroll
  for (int j = 0; j < 8; ++j) { r[j] = lo[j]; r[j+8] = hi[j]; }
  return r;
}
// B fragment from fp32 global: 16 contiguous reduction elems of this lane's weight row
static __device__ __forceinline__ v16bf load_b_f32(const float* __restrict__ p) {
  return cvt16(*(const float4*)(p), *(const float4*)(p+4),
               *(const float4*)(p+8), *(const float4*)(p+12));
}
// A fragment from bf16 global: K = {0..7} at p, {8..15} at p+8
static __device__ __forceinline__ v16bf load_a_bf16(const bf16_t* __restrict__ p) {
  return join16(*(const v8bf*)(p), *(const v8bf*)(p+8));
}

#define WMMA_BF16(A,B,C) __builtin_amdgcn_wmma_f32_16x16x32_bf16(false,(A),false,(B),(short)0,(C),false,false)

// ---------------- routing kernels ----------------
__global__ void k_init(float* __restrict__ out, int* __restrict__ perm,
                       float* __restrict__ pw, int* __restrict__ counts,
                       int* __restrict__ cursor) {
  int i = blockIdx.x * blockDim.x + threadIdx.x;
  int stride = gridDim.x * blockDim.x;
  for (int j = i; j < NTOK*HD; j += stride) out[j] = 0.f;
  for (int j = i; j < CAP; j += stride) { perm[j] = 0; pw[j] = 0.f; }
  if (i < NEXP) { counts[i] = 0; cursor[i] = 0; }
}

__global__ void k_count(const int* __restrict__ ids, int* __restrict__ counts) {
  int i = blockIdx.x * blockDim.x + threadIdx.x;
  if (i < NTOK*TOPK) atomicAdd(&counts[ids[i]], 1);
}

__global__ void k_offsets(const int* __restrict__ counts, int* __restrict__ offs,
                          int* __restrict__ nt, int2* __restrict__ tiles) {
  if (threadIdx.x != 0 || blockIdx.x != 0) return;
  int off = 0, n = 0;
  for (int e = 0; e < NEXP; ++e) {
    offs[e] = off;
    int t = (counts[e] + BM - 1) / BM;
    for (int j = 0; j < t; ++j) { tiles[n] = make_int2(e, off + j*BM); ++n; }
    off += t * BM;
  }
  offs[NEXP] = off;
  *nt = n;
}

__global__ void k_scatter(const int* __restrict__ ids, const float* __restrict__ wts,
                          const int* __restrict__ offs, int* __restrict__ cursor,
                          int* __restrict__ perm, float* __restrict__ pw) {
  int i = blockIdx.x * blockDim.x + threadIdx.x;
  if (i >= NTOK*TOPK) return;
  int e = ids[i];
  int slot = offs[e] + atomicAdd(&cursor[e], 1);
  perm[slot] = i / TOPK;       // token index
  pw[slot]   = wts[i];         // routing weight
}

// ---------------- GEMM1: gate+up via LDS-staged bf16 tiles, fused SwiGLU ----------------
__global__ __launch_bounds__(256) void k_gemm1(
    const float* __restrict__ x, const float* __restrict__ Wg, const float* __restrict__ Wu,
    const int* __restrict__ perm, const int2* __restrict__ tiles, const int* __restrict__ nt,
    bf16_t* __restrict__ hidden)
{
  if ((int)blockIdx.x >= *nt) return;              // uniform exit BEFORE any barrier
  __shared__ __align__(16) bf16_t sA[2*LDSA];
  __shared__ __align__(16) bf16_t sG[2*LDSB];
  __shared__ __align__(16) bf16_t sU[2*LDSB];

  const int2 td = tiles[blockIdx.x];
  const int e = td.x, row0 = td.y;
  const int tid = threadIdx.x;
  const int lane = tid & 31, wid = tid >> 5;
  const int wm = wid >> 2, wn = wid & 3;           // 2 x 4 waves
  const int lrow = lane & 15, lhi = lane >> 4;
  const int nb0 = blockIdx.y * BN;

  // --- staging assignments (256 threads cooperatively fill one k-slab) ---
  const int ar = tid >> 2, ak = (tid & 3) * 8;     // A: 64 rows x 32k, 8 floats/thread
  const int br = tid >> 1, bk = (tid & 1) * 16;    // B: 128 rows x 32k, 16 floats/thread/matrix
  const float* xrow = x  + (size_t)perm[row0 + ar] * HD + ak;
  const float* grow = Wg + ((size_t)e*ID + nb0 + br) * HD + bk;
  const float* urow = Wu + ((size_t)e*ID + nb0 + br) * HD + bk;

  float4 ra0, ra1, rg0, rg1, rg2, rg3, ru0, ru1, ru2, ru3;
#define GLOAD(kb)                                                              \
  { ra0 = *(const float4*)(xrow + (kb));     ra1 = *(const float4*)(xrow + (kb) + 4);  \
    rg0 = *(const float4*)(grow + (kb));     rg1 = *(const float4*)(grow + (kb) + 4);  \
    rg2 = *(const float4*)(grow + (kb) + 8); rg3 = *(const float4*)(grow + (kb) + 12); \
    ru0 = *(const float4*)(urow + (kb));     ru1 = *(const float4*)(urow + (kb) + 4);  \
    ru2 = *(const float4*)(urow + (kb) + 8); ru3 = *(const float4*)(urow + (kb) + 12); }
#define CVTSTORE(buf)                                                          \
  { *(v8bf*)&sA[(buf)*LDSA + ar*SAS + ak]     = cvt8(ra0, ra1);                \
    *(v8bf*)&sG[(buf)*LDSB + br*SAS + bk]     = cvt8(rg0, rg1);                \
    *(v8bf*)&sG[(buf)*LDSB + br*SAS + bk + 8] = cvt8(rg2, rg3);                \
    *(v8bf*)&sU[(buf)*LDSB + br*SAS + bk]     = cvt8(ru0, ru1);                \
    *(v8bf*)&sU[(buf)*LDSB + br*SAS + bk + 8] = cvt8(ru2, ru3); }

  // --- per-wave fragment offsets in LDS (elements) ---
  const int aoff0 = (wm*32 + lrow)      * SAS + lhi*16;  // A frag 0 (rows m..m+15)
  const int aoff1 = (wm*32 + 16 + lrow) * SAS + lhi*16;  // A frag 1
  const int boff0 = (wn*32 + lrow)      * SAS + lhi*16;  // B frag 0 (cols n..n+15)
  const int boff1 = (wn*32 + 16 + lrow) * SAS + lhi*16;  // B frag 1

  v8f accg[2][2], accu[2][2];
#pragma unroll
  for (int i = 0; i < 2; ++i)
#pragma unroll
    for (int j = 0; j < 2; ++j) { accg[i][j] = (v8f)0.f; accu[i][j] = (v8f)0.f; }

  // --- software pipeline: loads for step k+1 in flight while computing step k ---
  GLOAD(0);
  CVTSTORE(0);
  GLOAD(KC);
  __syncthreads();

  int cur = 0;
  for (int kb = 0; kb < HD; kb += KC) {
    // compute from LDS buffer `cur`
    const bf16_t* pA = sA + cur*LDSA;
    const bf16_t* pG = sG + cur*LDSB;
    const bf16_t* pU = sU + cur*LDSB;
    v16bf a0 = join16(*(const v8bf*)&pA[aoff0], *(const v8bf*)&pA[aoff0 + 8]);
    v16bf a1 = join16(*(const v8bf*)&pA[aoff1], *(const v8bf*)&pA[aoff1 + 8]);
    v16bf g0 = join16(*(const v8bf*)&pG[boff0], *(const v8bf*)&pG[boff0 + 8]);
    v16bf g1 = join16(*(const v8bf*)&pG[boff1], *(const v8bf*)&pG[boff1 + 8]);
    v16bf u0 = join16(*(const v8bf*)&pU[boff0], *(const v8bf*)&pU[boff0 + 8]);
    v16bf u1 = join16(*(const v8bf*)&pU[boff1], *(const v8bf*)&pU[boff1 + 8]);
    accg[0][0] = WMMA_BF16(a0, g0, accg[0][0]);
    accg[0][1] = WMMA_BF16(a0, g1, accg[0][1]);
    accg[1][0] = WMMA_BF16(a1, g0, accg[1][0]);
    accg[1][1] = WMMA_BF16(a1, g1, accg[1][1]);
    accu[0][0] = WMMA_BF16(a0, u0, accu[0][0]);
    accu[0][1] = WMMA_BF16(a0, u1, accu[0][1]);
    accu[1][0] = WMMA_BF16(a1, u0, accu[1][0]);
    accu[1][1] = WMMA_BF16(a1, u1, accu[1][1]);

    if (kb + KC < HD) {
      CVTSTORE(cur ^ 1);                 // waits on in-flight global loads, converts, stores
      if (kb + 2*KC < HD) GLOAD(kb + 2*KC);
    }
    __syncthreads();
    cur ^= 1;
  }

  // epilogue: hidden = silu(gate) * up, stored bf16 (C layout: M = r + 8*lhi, N = lrow)
  const int mwb = row0 + wm*32;
  const int nwb = nb0 + wn*32;
#pragma unroll
  for (int mf = 0; mf < 2; ++mf)
#pragma unroll
    for (int nf = 0; nf < 2; ++nf)
#pragma unroll
      for (int r = 0; r < 8; ++r) {
        const int row = mwb + mf*16 + lhi*8 + r;
        const int col = nwb + nf*16 + lrow;
        const float g = accg[mf][nf][r];
        const float u = accu[mf][nf][r];
        const float s = g / (1.f + __expf(-g));
        hidden[(size_t)row*ID + col] = (bf16_t)(s * u);
      }
}

// ---------------- GEMM2: down proj + weighted combine ----------------
__global__ __launch_bounds__(256) void k_gemm2(
    const float* __restrict__ Wd, const bf16_t* __restrict__ hidden,
    const int* __restrict__ perm, const float* __restrict__ pw,
    const int2* __restrict__ tiles, const int* __restrict__ nt,
    float* __restrict__ out)
{
  if ((int)blockIdx.x >= *nt) return;
  const int2 td = tiles[blockIdx.x];
  const int e = td.x, row0 = td.y;
  const int tid = threadIdx.x;
  const int lane = tid & 31, wid = tid >> 5;
  const int wm = wid >> 2, wn = wid & 3;
  const int lrow = lane & 15, lhi = lane >> 4;
  const int mbase = row0 + wm*32;
  const int nbase = blockIdx.y*BN + wn*32;

  const bf16_t* ha0 = hidden + (size_t)(mbase + lrow     )*ID + lhi*16;
  const bf16_t* ha1 = hidden + (size_t)(mbase + 16 + lrow)*ID + lhi*16;
  const float*  b0  = Wd + ((size_t)e*HD + nbase + lrow     )*ID + lhi*16;
  const float*  b1  = Wd + ((size_t)e*HD + nbase + 16 + lrow)*ID + lhi*16;

  v8f acc[2][2];
#pragma unroll
  for (int i = 0; i < 2; ++i)
#pragma unroll
    for (int j = 0; j < 2; ++j) acc[i][j] = (v8f)0.f;

  for (int kb = 0; kb < ID; kb += KC) {
    __builtin_prefetch(b0 + kb + 64, 0, 1);
    v16bf a0 = load_a_bf16(ha0 + kb);
    v16bf a1 = load_a_bf16(ha1 + kb);
    v16bf w0 = load_b_f32(b0 + kb);
    v16bf w1 = load_b_f32(b1 + kb);
    acc[0][0] = WMMA_BF16(a0, w0, acc[0][0]);
    acc[0][1] = WMMA_BF16(a0, w1, acc[0][1]);
    acc[1][0] = WMMA_BF16(a1, w0, acc[1][0]);
    acc[1][1] = WMMA_BF16(a1, w1, acc[1][1]);
  }

  // combine: out[token, h] += weight * val  (pad rows have weight 0)
#pragma unroll
  for (int mf = 0; mf < 2; ++mf)
#pragma unroll
    for (int r = 0; r < 8; ++r) {
      const int slot = mbase + mf*16 + lhi*8 + r;
      const int tok  = perm[slot];
      const float w  = pw[slot];
#pragma unroll
      for (int nf = 0; nf < 2; ++nf) {
        const int col = nbase + nf*16 + lrow;
        atomicAdd(&out[(size_t)tok*HD + col], acc[mf][nf][r] * w);
      }
    }
}

// ---------------- launch ----------------
extern "C" void kernel_launch(void* const* d_in, const int* in_sizes, int n_in,
                              void* d_out, int out_size, void* d_ws, size_t ws_size,
                              hipStream_t stream) {
  const float* x   = (const float*)d_in[0];
  const int*   ids = (const int*)  d_in[1];
  const float* wts = (const float*)d_in[2];
  const float* Wg  = (const float*)d_in[3];
  const float* Wu  = (const float*)d_in[4];
  const float* Wd  = (const float*)d_in[5];
  float* out = (float*)d_out;

  char* ws = (char*)d_ws;
  int*    counts = (int*)   (ws + OFF_COUNTS);
  int*    cursor = (int*)   (ws + OFF_CURSOR);
  int*    offs   = (int*)   (ws + OFF_OFFS);
  int*    nt     = (int*)   (ws + OFF_NT);
  int2*   tiles  = (int2*)  (ws + OFF_TILES);
  int*    perm   = (int*)   (ws + OFF_PERM);
  float*  pw     = (float*) (ws + OFF_PW);
  bf16_t* hidden = (bf16_t*)(ws + OFF_HID);

  k_init   <<<2048, 256, 0, stream>>>(out, perm, pw, counts, cursor);
  k_count  <<<(NTOK*TOPK + 255)/256, 256, 0, stream>>>(ids, counts);
  k_offsets<<<1, 1, 0, stream>>>(counts, offs, nt, tiles);
  k_scatter<<<(NTOK*TOPK + 255)/256, 256, 0, stream>>>(ids, wts, offs, cursor, perm, pw);
  k_gemm1  <<<dim3(MAXT, ID/BN), 256, 0, stream>>>(x, Wg, Wu, perm, tiles, nt, hidden);
  k_gemm2  <<<dim3(MAXT, HD/BN), 256, 0, stream>>>(Wd, hidden, perm, pw, tiles, nt, out);
}
